// QKProjection_77884936945984
// MI455X (gfx1250) — matrix operand order repaired and checked by reading, not compiled, not run on hardware
//
#include <hip/hip_runtime.h>
#include <hip/hip_bf16.h>
#include <math.h>

// ---------------------------------------------------------------------------
// Causal linear attention, flattened from the reference's sequential chunk
// scan into the fully parallel block form:
//   out = Q @ Mp^T + tril(Q K^T) @ K,  row-scaled by 1/n_t.
// ~157 GFLOP of f16 WMMA (f32 accumulate). Q/K/Mp are pre-converted to f16
// (one streaming pass), shared B tiles staged in LDS (padded rows, no bank
// conflicts). A sched_barrier keeps each group's 16 ds_load_b128 hoisted
// ahead of its 8 WMMAs so the matrix pipe issues back-to-back under partial
// dscnt waits instead of a full drain per WMMA.
// Inputs (64MB) are L2-resident on MI455X (192MB L2).
// ---------------------------------------------------------------------------

#define T_LEN  8192
#define DIMN   1024
#define CHUNK  128
#define NCH    (T_LEN / CHUNK)          // 64
#define NPAIR  (NCH * (NCH + 1) / 2)    // 2080
#define NORM_PERSISTENT 1024.0f
#define TP     136                      // padded LDS tile row (f16 elems)

typedef __attribute__((ext_vector_type(16))) _Float16 v16h;
typedef __attribute__((ext_vector_type(8)))  _Float16 h8;
typedef __attribute__((ext_vector_type(4)))  _Float16 h4;
typedef __attribute__((ext_vector_type(8)))  float    v8f;

__device__ __forceinline__ void sched_fence() {
#if __has_builtin(__builtin_amdgcn_sched_barrier)
  __builtin_amdgcn_sched_barrier(0);   // nothing crosses this point
#endif
}

// ---- CDNA5 WMMA fragment helpers (wave32; contiguous k order chosen here, same for A and B) ----------
// A 16x32 f16: lane -> row (lane&15); VGPR i holds K pair
//   k = 2i + 16*hf  => 16 contiguous halfwords at k0+16*hf (two 8-halfword
//   chunks at k0+16*hf and k0+16*hf+8).
// B 32x16 f16 from a transposed tile T (T[n][k] = B[k][n]): lane -> col n;
//   VGPR i holds pair k = 2i + 16*hf => 16 contiguous halfwords at k0+16*hf.

__device__ __forceinline__ v16h load_afrag(const _Float16* __restrict__ base,
                                           int hf) {
  const h8 x0 = *(const h8*)(base + 16 * hf);       // k0+16hf  .. +7
  const h8 x1 = *(const h8*)(base + 16 * hf + 8);   // k0+16hf+8.. +7
  v16h r;
#pragma unroll
  for (int i = 0; i < 8; ++i) { r[i] = x0[i]; r[8 + i] = x1[i]; }
  return r;
}

__device__ __forceinline__ v16h load_bfrag(const _Float16* __restrict__ p) {
  // p = &tile[n][k0 + 16*hf]; 16 contiguous halfwords (2 x ds_load_b128)
  const h8 x0 = *(const h8*)p;
  const h8 x1 = *(const h8*)(p + 8);
  v16h r;
#pragma unroll
  for (int i = 0; i < 8; ++i) { r[i] = x0[i]; r[8 + i] = x1[i]; }
  return r;
}

__device__ __forceinline__ v8f wmma_f16(v16h a, v16h b, v8f c) {
  return __builtin_amdgcn_wmma_f32_16x16x32_f16(false, a, false, b,
                                                (short)0, c, false, false);
}

// Cooperative stage of a 128x128 f16 tile into LDS (row-padded to TP).
// 256 threads: thread -> (row = tid>>1, 64-elem half-row), 8 x 16B each.
__device__ __forceinline__ void stage_tile(_Float16* __restrict__ tile,
                                           const _Float16* __restrict__ src,
                                           int ld, int tid) {
  const int row  = tid >> 1;
  const int seg  = (tid & 1) * 64;
  const _Float16* s = src + (size_t)row * ld + seg;
  _Float16* d = tile + row * TP + seg;
#pragma unroll
  for (int i = 0; i < 8; ++i) {
    *(h8*)(d + i * 8) = *(const h8*)(s + i * 8);
  }
}

// 8 B-fragments loaded up front (16 ds_load_b128 kept in flight by the
// sched_barrier), then 8 WMMAs issued back-to-back against a shared A frag.
__device__ __forceinline__ void mma_group(const _Float16* __restrict__ tile,
                                          int k0, int lo, int hf,
                                          const v16h a, v8f acc[8]) {
  v16h b[8];
#pragma unroll
  for (int nt = 0; nt < 8; ++nt)
    b[nt] = load_bfrag(&tile[(nt * 16 + lo) * TP + k0 + 16 * hf]);
  sched_fence();   // keep all ds_loads above, all WMMAs below
#pragma unroll
  for (int nt = 0; nt < 8; ++nt)
    acc[nt] = wmma_f16(a, b[nt], acc[nt]);
}

// ---------------------------------------------------------------------------
// Kernel 1: per-row sum of squares of K (one wave per row)
// ---------------------------------------------------------------------------
__global__ __launch_bounds__(256) void rowsumsq_kernel(const float* __restrict__ K,
                                                       float* __restrict__ rowsq) {
  const int w    = threadIdx.x >> 5;
  const int lane = threadIdx.x & 31;
  const int row  = blockIdx.x * 8 + w;
  const float* p = K + (size_t)row * DIMN;
  float s = 0.0f;
#pragma unroll
  for (int i = 0; i < DIMN / 32; ++i) {
    const float v = p[lane + 32 * i];
    s += v * v;
  }
#pragma unroll
  for (int o = 16; o > 0; o >>= 1) s += __shfl_xor(s, o, 32);
  if (lane == 0) rowsq[row] = s;
}

// ---------------------------------------------------------------------------
// Kernel 2: single-block inclusive scan -> norms[t] = 1024 + cumsum(rowsq)
// ---------------------------------------------------------------------------
__global__ __launch_bounds__(1024) void scan_kernel(const float* __restrict__ rowsq,
                                                    float* __restrict__ norms) {
  __shared__ float sp[1024];
  const int tid = threadIdx.x;
  float part = 0.0f;
  float loc[8];
#pragma unroll
  for (int i = 0; i < 8; ++i) {
    loc[i] = rowsq[tid * 8 + i];
    part += loc[i];
  }
  sp[tid] = part;
  __syncthreads();
  for (int off = 1; off < 1024; off <<= 1) {
    const float add = (tid >= off) ? sp[tid - off] : 0.0f;
    __syncthreads();
    sp[tid] += add;
    __syncthreads();
  }
  float run = NORM_PERSISTENT + (sp[tid] - part);  // exclusive prefix
#pragma unroll
  for (int i = 0; i < 8; ++i) {
    run += loc[i];
    norms[tid * 8 + i] = run;
  }
}

// ---------------------------------------------------------------------------
// Kernel 3: flat f32 -> f16 convert (Qh, Kh, Mh), 4 elems/thread
// ---------------------------------------------------------------------------
__global__ __launch_bounds__(256) void cvt_f16_kernel(const float* __restrict__ src,
                                                      _Float16* __restrict__ dst,
                                                      int n4) {
  const int idx = blockIdx.x * 256 + threadIdx.x;
  if (idx < n4) {
    const float4 v = ((const float4*)src)[idx];
    h4 o;
    o[0] = (_Float16)v.x; o[1] = (_Float16)v.y;
    o[2] = (_Float16)v.z; o[3] = (_Float16)v.w;
    ((h4*)dst)[idx] = o;
  }
}

// ---------------------------------------------------------------------------
// Kernel 4: KT[f][t] = (f16) K[t][f]   (LDS-tiled transpose + downconvert)
// ---------------------------------------------------------------------------
__global__ __launch_bounds__(256) void transpose_kernel(const float* __restrict__ K,
                                                        _Float16* __restrict__ KT) {
  __shared__ float tile[32][33];
  const int tx = threadIdx.x, ty = threadIdx.y;      // 32 x 8
  const int t0 = blockIdx.x * 32, f0 = blockIdx.y * 32;
#pragma unroll
  for (int i = 0; i < 4; ++i) {
    const int rr = ty + i * 8;
    tile[rr][tx] = K[(size_t)(t0 + rr) * DIMN + f0 + tx];
  }
  __syncthreads();
#pragma unroll
  for (int i = 0; i < 4; ++i) {
    const int fr = ty + i * 8;
    KT[(size_t)(f0 + fr) * T_LEN + t0 + tx] = (_Float16)tile[tx][fr];
  }
}

// ---------------------------------------------------------------------------
// Kernel 5: score blocks  S(c,j) = Q_c K_j^T  (causal mask on diagonal block)
// One workgroup per (c,j); wave w owns rows [16w,16w+16); B tiles via LDS.
// ---------------------------------------------------------------------------
__global__ __launch_bounds__(256) void s_blocks_kernel(const _Float16* __restrict__ Qh,
                                                       const _Float16* __restrict__ Kh,
                                                       _Float16* __restrict__ S) {
  __shared__ _Float16 tile[CHUNK * TP];

  const int p = blockIdx.x;
  int c = (int)((sqrtf(8.0f * (float)p + 1.0f) - 1.0f) * 0.5f);
  while ((c + 1) * (c + 2) / 2 <= p) ++c;
  while (c * (c + 1) / 2 > p) --c;
  const int j = p - c * (c + 1) / 2;

  const int tid  = threadIdx.x;
  const int w    = tid >> 5;
  const int lane = tid & 31;
  const int lo   = lane & 15;
  const int hf   = lane >> 4;

  v8f acc[8];
#pragma unroll
  for (int nt = 0; nt < 8; ++nt) acc[nt] = {};

  const _Float16* qrow = Qh + (size_t)(c * CHUNK + 16 * w + lo) * DIMN;
#pragma unroll 1
  for (int e0 = 0; e0 < DIMN; e0 += 128) {
    // tile[n][e] = Kh[j*128+n][e0+e]  (B[e][n] = K[n][e])
    stage_tile(tile, Kh + (size_t)(j * CHUNK) * DIMN + e0, DIMN, tid);
    __syncthreads();
#pragma unroll
    for (int ee = 0; ee < 128; ee += 32) {
      const v16h a = load_afrag(qrow + e0 + ee, hf);
      mma_group(tile, ee, lo, hf, a, acc);
    }
    __syncthreads();
  }

  if (j == c) {  // causal: keep s <= t  (n <= m within diagonal block)
#pragma unroll
    for (int nt = 0; nt < 8; ++nt) {
      const int n = nt * 16 + lo;
#pragma unroll
      for (int r = 0; r < 8; ++r) {
        const int m = 16 * w + r + 8 * hf;
        if (n > m) acc[nt][r] = 0.0f;
      }
    }
  }

  _Float16* sbase = S + (size_t)p * CHUNK * CHUNK;
#pragma unroll
  for (int nt = 0; nt < 8; ++nt) {
    const int n = nt * 16 + lo;
#pragma unroll
    for (int r = 0; r < 8; ++r) {
      const int m = 16 * w + r + 8 * hf;
      sbase[m * CHUNK + n] = (_Float16)acc[nt][r];
    }
  }
}

// ---------------------------------------------------------------------------
// Kernel 6: out tile (c,colg) = [ Q_c @ Mp^T + sum_{j<=c} S(c,j) @ K_j ] / n_t
// grid (8 col groups of 128, 64 chunks); wave w -> rows [16w,16w+16).
// ---------------------------------------------------------------------------
__global__ __launch_bounds__(256) void out_kernel(const _Float16* __restrict__ Qh,
                                                  const _Float16* __restrict__ Mh,
                                                  const _Float16* __restrict__ S,
                                                  const _Float16* __restrict__ KT,
                                                  const float* __restrict__ norms,
                                                  float* __restrict__ out) {
  __shared__ _Float16 tile[CHUNK * TP];
  __shared__ float sinv[CHUNK];

  const int colg = blockIdx.x;       // 0..7
  const int c    = blockIdx.y;       // 0..63
  const int tid  = threadIdx.x;
  const int w    = tid >> 5;
  const int lane = tid & 31;
  const int lo   = lane & 15;
  const int hf   = lane >> 4;
  const int colbase = colg * 128;

  if (tid < CHUNK) sinv[tid] = 1.0f / fmaxf(norms[c * CHUNK + tid], 1e-8f);

  v8f acc[8];
#pragma unroll
  for (int nt = 0; nt < 8; ++nt) acc[nt] = {};

  const _Float16* qrow = Qh + (size_t)(c * CHUNK + 16 * w + lo) * DIMN;

  // Part A: Q_c @ Mp^T   (tile[d][e] = Mh[colbase+d][e0+e])
#pragma unroll 1
  for (int e0 = 0; e0 < DIMN; e0 += 128) {
    stage_tile(tile, Mh + (size_t)colbase * DIMN + e0, DIMN, tid);
    __syncthreads();
#pragma unroll
    for (int ee = 0; ee < 128; ee += 32) {
      const v16h a = load_afrag(qrow + e0 + ee, hf);
      mma_group(tile, ee, lo, hf, a, acc);
    }
    __syncthreads();
  }

  // Part B: sum_j S(c,j) @ K_j   (tile[n][k] = KT[colbase+n][j*128+k])
  const size_t pbase = (size_t)c * (c + 1) / 2;
#pragma unroll 1
  for (int j = 0; j <= c; ++j) {
    stage_tile(tile, KT + (size_t)colbase * T_LEN + j * CHUNK, T_LEN, tid);
    __syncthreads();
    const _Float16* sb =
        S + (pbase + j) * (size_t)(CHUNK * CHUNK) + (size_t)(16 * w + lo) * CHUNK;
#pragma unroll
    for (int kk = 0; kk < CHUNK; kk += 32) {
      const v16h a = load_afrag(sb + kk, hf);
      mma_group(tile, kk, lo, hf, a, acc);
    }
    __syncthreads();
  }

  // Epilogue: row scale + store f32
  float* obase = out + (size_t)(c * CHUNK) * DIMN + colbase;
#pragma unroll
  for (int nt = 0; nt < 8; ++nt) {
    const int n = nt * 16 + lo;
#pragma unroll
    for (int r = 0; r < 8; ++r) {
      const int m = 16 * w + r + 8 * hf;
      obase[(size_t)m * DIMN + n] = acc[nt][r] * sinv[m];
    }
  }
}

// ---------------------------------------------------------------------------
// Workspace layout (bytes):
//   [0,        32768)   rowsq  8192 f32
//   [32768,    65536)   norms  8192 f32
//   @65536              KT  1024x8192 f16   (16 MiB)
//   +16MiB              Qh  8192x1024 f16   (16 MiB)
//   +32MiB              Kh  8192x1024 f16   (16 MiB)
//   +48MiB              Mh  1024x1024 f16   ( 2 MiB)
//   +50MiB              S   2080x128x128 f16 (~65 MiB)
// Total ~115.1 MiB.
// ---------------------------------------------------------------------------
extern "C" void kernel_launch(void* const* d_in, const int* in_sizes, int n_in,
                              void* d_out, int out_size, void* d_ws, size_t ws_size,
                              hipStream_t stream) {
  const float* Q  = (const float*)d_in[0];
  const float* K  = (const float*)d_in[1];
  const float* Mp = (const float*)d_in[2];
  float* out = (float*)d_out;

  char* ws = (char*)d_ws;
  const size_t MB = 1024 * 1024;
  float*    rowsq = (float*)ws;
  float*    norms = (float*)(ws + 32768);
  _Float16* KT    = (_Float16*)(ws + 65536);
  _Float16* Qh    = (_Float16*)(ws + 65536 + 16 * MB);
  _Float16* Kh    = (_Float16*)(ws + 65536 + 32 * MB);
  _Float16* Mh    = (_Float16*)(ws + 65536 + 48 * MB);
  _Float16* S     = (_Float16*)(ws + 65536 + 50 * MB);

  rowsumsq_kernel<<<dim3(T_LEN / 8), dim3(256), 0, stream>>>(K, rowsq);
  scan_kernel<<<dim3(1), dim3(1024), 0, stream>>>(rowsq, norms);

  const int nQ4 = T_LEN * DIMN / 4, nM4 = DIMN * DIMN / 4;
  cvt_f16_kernel<<<dim3((nQ4 + 255) / 256), dim3(256), 0, stream>>>(Q, Qh, nQ4);
  cvt_f16_kernel<<<dim3((nQ4 + 255) / 256), dim3(256), 0, stream>>>(K, Kh, nQ4);
  cvt_f16_kernel<<<dim3((nM4 + 255) / 256), dim3(256), 0, stream>>>(Mp, Mh, nM4);
  transpose_kernel<<<dim3(T_LEN / 32, DIMN / 32), dim3(32, 8), 0, stream>>>(K, KT);

  s_blocks_kernel<<<dim3(NPAIR), dim3(256), 0, stream>>>(Qh, Kh, S);
  out_kernel<<<dim3(8, NCH), dim3(256), 0, stream>>>(Qh, Mh, S, KT, norms, out);
}
